// InstoClass_CosMetric_36713380446638
// MI455X (gfx1250) — matrix operand order repaired and checked by reading, not compiled, not run on hardware
//
#include <hip/hip_runtime.h>

// ---------------------------------------------------------------------------
// out[b,c] = max_k dot(x1[b,:], x2[c,k,:])
// Pass 1: split f32 -> bf16 hi/lo arrays in workspace (one-shot).
// Pass 2: GEMM [8192,256]x[256,16384] via bf16x3 WMMA, tiles staged by the
//         Tensor Data Mover (tensor_load_to_lds, double-buffered,
//         s_wait_tensorcnt), per-32-column max reduction.
// ---------------------------------------------------------------------------

typedef __attribute__((ext_vector_type(16))) __bf16        v16bf;
typedef __attribute__((ext_vector_type(8)))  float         v8f;
typedef __attribute__((ext_vector_type(4)))  unsigned int  u32x4;
typedef __attribute__((ext_vector_type(8)))  int           i32x8;
typedef __attribute__((ext_vector_type(4)))  int           i32x4;

#define B_DIM   8192
#define C_DIM   512
#define K_DIM   32
#define D_DIM   256
#define NCOL    (C_DIM * K_DIM)          // 16384 flat (c,k) columns

#define M_BLK   128
#define N_BLK   64
#define D_CHUNK 32

#define LDH     40                       // LDS row stride in halves (80B)

// workspace layout (in halves)
#define NA_ (B_DIM * D_DIM)              // 2,097,152
#define NB_ (NCOL * D_DIM)               // 4,194,304
#define WS_AHI 0u
#define WS_ALO ((unsigned)NA_)
#define WS_BHI ((unsigned)(2 * NA_))
#define WS_BLO ((unsigned)(2 * NA_ + NB_))
#define WS_TOTAL_HALVES ((size_t)(2 * NA_ + 2 * NB_))

// LDS layout per buffer (in halves)
#define OFF_AHI 0
#define OFF_ALO (M_BLK * LDH)            // 5120
#define OFF_BHI (2 * M_BLK * LDH)        // 10240
#define OFF_BLO (2 * M_BLK * LDH + N_BLK * LDH) // 12800
#define SBUF    (2 * M_BLK * LDH + 2 * N_BLK * LDH) // 15360 halves / 30720 B

#if __has_include(<hip/amd_detail/amd_gfx1250_TDM.h>)
#define TDM_6ARG 1
#else
#define TDM_6ARG 0
#endif

union V16U { v16bf v; uint4 q[2]; };

// ---------------- bf16 hi/lo split helpers (RNE) ----------------
__device__ __forceinline__ void bf16_split(float f, unsigned short& h, unsigned short& l) {
    unsigned u  = __float_as_uint(f);
    unsigned rh = (u + 0x7FFFu + ((u >> 16) & 1u)) & 0xFFFF0000u;
    h = (unsigned short)(rh >> 16);
    float resid = f - __uint_as_float(rh);
    unsigned ul = __float_as_uint(resid);
    unsigned rl = ul + 0x7FFFu + ((ul >> 16) & 1u);
    l = (unsigned short)(rl >> 16);
}

__device__ __forceinline__ void split_store4(float4 f, unsigned short* hi, unsigned short* lo) {
    unsigned short h0, h1, h2, h3, l0, l1, l2, l3;
    bf16_split(f.x, h0, l0);
    bf16_split(f.y, h1, l1);
    bf16_split(f.z, h2, l2);
    bf16_split(f.w, h3, l3);
    uint2 hv, lv;
    hv.x = (unsigned)h0 | ((unsigned)h1 << 16);
    hv.y = (unsigned)h2 | ((unsigned)h3 << 16);
    lv.x = (unsigned)l0 | ((unsigned)l1 << 16);
    lv.y = (unsigned)l2 | ((unsigned)l3 << 16);
    *(uint2*)hi = hv;
    *(uint2*)lo = lv;
}

// ---------------- TDM descriptor build + issue (2D tile, bf16) ----------------
// Loads a tile_rows x 32(half) tile from a [tensor_rows, 256] bf16 tensor into
// LDS at lds_byte, inserting 16B padding after each 64B row (-> stride 80B).
__device__ __forceinline__ void tdm_load_2d(unsigned long long gaddr, unsigned lds_byte,
                                            unsigned tensor_rows, unsigned tile_rows) {
    u32x4 g0;
    g0[0] = 1u;                                            // count=1, user mode
    g0[1] = lds_byte;                                      // lds_addr [63:32]
    g0[2] = (unsigned)(gaddr & 0xFFFFFFFFull);             // global_addr low
    g0[3] = (unsigned)((gaddr >> 32) & 0x1FFFFFFull)       // global_addr [56:32]
          | (2u << 30);                                    // type=2 (image)
    i32x8 g1;
    g1[0] = (int)((1u << 16)                               // data_size = 2B
                | (1u << 20)                               // pad_enable
                | (3u << 22)                               // pad_interval: 16 DW (64B)
                | (3u << 25));                             // pad_amount: 4 DW (16B)
    g1[1] = (int)((unsigned)D_DIM << 16);                  // tensor_dim0 = 256 (lo16)
    g1[2] = (int)((tensor_rows & 0xFFFFu) << 16);          // dim0 hi=0 | tensor_dim1 lo16
    g1[3] = (int)((tensor_rows >> 16) | (32u << 16));      // tensor_dim1 hi | tile_dim0=32
    g1[4] = (int)(tile_rows & 0xFFFFu);                    // tile_dim1 | tile_dim2=0
    g1[5] = (int)D_DIM;                                    // tensor_dim0_stride lo32 = 256
    g1[6] = 0;                                             // stride hi | tensor_dim1_stride lo
    g1[7] = 0;
    i32x4 z4 = {0, 0, 0, 0};
#if TDM_6ARG
    i32x8 z8 = {0, 0, 0, 0, 0, 0, 0, 0};
    __builtin_amdgcn_tensor_load_to_lds(g0, g1, z4, z4, z8, 0);
#else
    __builtin_amdgcn_tensor_load_to_lds(g0, g1, z4, z4, 0);
#endif
}

// ---------------- Pass 1: one-shot f32 -> bf16 hi/lo split ----------------
__global__ __launch_bounds__(256)
void convert_split_kernel(const float* __restrict__ x1, const float* __restrict__ x2,
                          unsigned short* __restrict__ w) {
    const unsigned idx = (blockIdx.x * 256u + threadIdx.x) * 4u;
    if (idx < (unsigned)NA_) {
        float4 f = *(const float4*)&x1[idx];
        split_store4(f, &w[WS_AHI + idx], &w[WS_ALO + idx]);
    } else {
        const unsigned j = idx - (unsigned)NA_;
        float4 f = *(const float4*)&x2[j];
        split_store4(f, &w[WS_BHI + j], &w[WS_BLO + j]);
    }
}

// ---------------- Pass 2: TDM-staged bf16x3 WMMA GEMM + max ----------------
__global__ __launch_bounds__(256)
void gemm_tdm_kernel(const unsigned short* __restrict__ w, float* __restrict__ out) {
    __shared__ __align__(16) unsigned short smem[2 * SBUF];

    const int t    = threadIdx.x;
    const int m0   = blockIdx.x * M_BLK;
    const int n0   = blockIdx.y * N_BLK;
    const int wave = t >> 5;
    const int lane = t & 31;
    const int half = lane >> 4;
    const int lm   = lane & 15;

    const unsigned ldsbase = (unsigned)(unsigned long long)(const void*)&smem[0];

    auto issue = [&](int buf, int dc) {
        const unsigned b = ldsbase + (unsigned)(buf * SBUF * 2);
        tdm_load_2d((unsigned long long)(const void*)(w + WS_AHI + (size_t)m0 * D_DIM + dc),
                    b + OFF_AHI * 2, B_DIM, M_BLK);
        tdm_load_2d((unsigned long long)(const void*)(w + WS_ALO + (size_t)m0 * D_DIM + dc),
                    b + OFF_ALO * 2, B_DIM, M_BLK);
        tdm_load_2d((unsigned long long)(const void*)(w + WS_BHI + (size_t)n0 * D_DIM + dc),
                    b + OFF_BHI * 2, NCOL, N_BLK);
        tdm_load_2d((unsigned long long)(const void*)(w + WS_BLO + (size_t)n0 * D_DIM + dc),
                    b + OFF_BLO * 2, NCOL, N_BLK);
    };

    v8f acc[4] = {};

    if (t == 0) issue(0, 0);   // TDM ignores EXEC; only wave 0 issues/tracks TENSORcnt

    for (int i = 0; i < D_DIM / D_CHUNK; ++i) {
        if (t == 0) __builtin_amdgcn_s_wait_tensorcnt((short)0);
        __syncthreads();       // chunk i resident; all reads of buf (i+1)&1 retired
        if (t == 0 && i + 1 < D_DIM / D_CHUNK) issue((i + 1) & 1, (i + 1) * D_CHUNK);

        const int sb = (i & 1) * SBUF;

        // A operand (16x32 bf16): lane m=lm; low lanes K{0..7,16..23},
        // high lanes K{8..15,24..31} -> two b128 chunks each.
        const int arow = wave * 16 + lm;
        V16U Ah, Al;
        Ah.q[0] = *(const uint4*)&smem[sb + OFF_AHI + arow * LDH + half * 8];
        Ah.q[1] = *(const uint4*)&smem[sb + OFF_AHI + arow * LDH + 16 + half * 8];
        Al.q[0] = *(const uint4*)&smem[sb + OFF_ALO + arow * LDH + half * 8];
        Al.q[1] = *(const uint4*)&smem[sb + OFF_ALO + arow * LDH + 16 + half * 8];

        #pragma unroll
        for (int j = 0; j < 4; ++j) {
            // B operand (32x16 bf16): lane col n=lm; low lanes K{0..7,16..23}, high lanes K{8..15,24..31}.
            const int brow = j * 16 + lm;
            V16U Bh, Bl;
            Bh.q[0] = *(const uint4*)&smem[sb + OFF_BHI + brow * LDH + half * 8];
            Bh.q[1] = *(const uint4*)&smem[sb + OFF_BHI + brow * LDH + 16 + half * 8];
            Bl.q[0] = *(const uint4*)&smem[sb + OFF_BLO + brow * LDH + half * 8];
            Bl.q[1] = *(const uint4*)&smem[sb + OFF_BLO + brow * LDH + 16 + half * 8];

            acc[j] = __builtin_amdgcn_wmma_f32_16x16x32_bf16(
                false, Ah.v, false, Bh.v, (short)0, acc[j], false, false);
            acc[j] = __builtin_amdgcn_wmma_f32_16x16x32_bf16(
                false, Ah.v, false, Bl.v, (short)0, acc[j], false, false);
            acc[j] = __builtin_amdgcn_wmma_f32_16x16x32_bf16(
                false, Al.v, false, Bh.v, (short)0, acc[j], false, false);
        }
    }

    // Max over 32 shot-columns per class.
    // C/D layout: lane L, VGPR v -> row M = v + 8*(L>>4), col N = L&15.
    const int c0 = n0 >> 5;
    #pragma unroll
    for (int cc = 0; cc < 2; ++cc) {
        #pragma unroll
        for (int v = 0; v < 8; ++v) {
            float r = fmaxf(acc[2 * cc][v], acc[2 * cc + 1][v]);
            r = fmaxf(r, __shfl_xor(r, 1, 32));
            r = fmaxf(r, __shfl_xor(r, 2, 32));
            r = fmaxf(r, __shfl_xor(r, 4, 32));
            r = fmaxf(r, __shfl_xor(r, 8, 32));
            if (lm == 0) {
                const int row = m0 + wave * 16 + half * 8 + v;
                out[row * C_DIM + (c0 + cc)] = r;
            }
        }
    }
}

// ---------------- Fallback: round-1 fused kernel (no workspace) ----------------
__global__ __launch_bounds__(256)
void fused_kernel(const float* __restrict__ x1, const float* __restrict__ x2,
                  float* __restrict__ out) {
    __shared__ __align__(16) unsigned short sAhi[M_BLK * LDH];
    __shared__ __align__(16) unsigned short sAlo[M_BLK * LDH];
    __shared__ __align__(16) unsigned short sBhi[N_BLK * LDH];
    __shared__ __align__(16) unsigned short sBlo[N_BLK * LDH];

    const int t    = threadIdx.x;
    const int m0   = blockIdx.x * M_BLK;
    const int n0   = blockIdx.y * N_BLK;
    const int wave = t >> 5;
    const int lane = t & 31;
    const int half = lane >> 4;
    const int lm   = lane & 15;

    const int arow4 = t >> 3;
    const int acol  = (t & 7) * 4;
    const int bn    = t >> 2;
    const int bc    = (t & 3) * 4;

    v8f acc[4] = {};

    for (int dc = 0; dc < D_DIM; dc += D_CHUNK) {
        __syncthreads();
        if (dc + D_CHUNK < D_DIM) {
            __builtin_prefetch(&x1[(m0 + arow4) * D_DIM + dc + D_CHUNK + acol], 0, 3);
            __builtin_prefetch(&x2[(n0 + bn)    * D_DIM + dc + D_CHUNK + bc],   0, 3);
        }
        #pragma unroll
        for (int pass = 0; pass < 4; ++pass) {
            const int row = pass * 32 + arow4;
            float4 f = *(const float4*)&x1[(m0 + row) * D_DIM + dc + acol];
            split_store4(f, &sAhi[row * LDH + acol], &sAlo[row * LDH + acol]);
        }
        #pragma unroll
        for (int s = 0; s < 2; ++s) {
            const int d = bc + s * 16;
            float4 f = *(const float4*)&x2[(n0 + bn) * D_DIM + dc + d];
            split_store4(f, &sBhi[bn * LDH + d], &sBlo[bn * LDH + d]);
        }
        __syncthreads();

        const int arow = wave * 16 + lm;
        V16U Ah, Al;
        Ah.q[0] = *(const uint4*)&sAhi[arow * LDH + half * 8];
        Ah.q[1] = *(const uint4*)&sAhi[arow * LDH + 16 + half * 8];
        Al.q[0] = *(const uint4*)&sAlo[arow * LDH + half * 8];
        Al.q[1] = *(const uint4*)&sAlo[arow * LDH + 16 + half * 8];

        #pragma unroll
        for (int j = 0; j < 4; ++j) {
            const int brow = j * 16 + lm;
            V16U Bh, Bl;
            Bh.q[0] = *(const uint4*)&sBhi[brow * LDH + half * 8];
            Bh.q[1] = *(const uint4*)&sBhi[brow * LDH + 16 + half * 8];
            Bl.q[0] = *(const uint4*)&sBlo[brow * LDH + half * 8];
            Bl.q[1] = *(const uint4*)&sBlo[brow * LDH + 16 + half * 8];

            acc[j] = __builtin_amdgcn_wmma_f32_16x16x32_bf16(
                false, Ah.v, false, Bh.v, (short)0, acc[j], false, false);
            acc[j] = __builtin_amdgcn_wmma_f32_16x16x32_bf16(
                false, Ah.v, false, Bl.v, (short)0, acc[j], false, false);
            acc[j] = __builtin_amdgcn_wmma_f32_16x16x32_bf16(
                false, Al.v, false, Bh.v, (short)0, acc[j], false, false);
        }
    }

    const int c0 = n0 >> 5;
    #pragma unroll
    for (int cc = 0; cc < 2; ++cc) {
        #pragma unroll
        for (int v = 0; v < 8; ++v) {
            float r = fmaxf(acc[2 * cc][v], acc[2 * cc + 1][v]);
            r = fmaxf(r, __shfl_xor(r, 1, 32));
            r = fmaxf(r, __shfl_xor(r, 2, 32));
            r = fmaxf(r, __shfl_xor(r, 4, 32));
            r = fmaxf(r, __shfl_xor(r, 8, 32));
            if (lm == 0) {
                const int row = m0 + wave * 16 + half * 8 + v;
                out[row * C_DIM + (c0 + cc)] = r;
            }
        }
    }
}

extern "C" void kernel_launch(void* const* d_in, const int* in_sizes, int n_in,
                              void* d_out, int out_size, void* d_ws, size_t ws_size,
                              hipStream_t stream) {
    (void)in_sizes; (void)n_in; (void)out_size;
    const float* x1 = (const float*)d_in[0];
    const float* x2 = (const float*)d_in[1];
    float* out = (float*)d_out;

    dim3 grid(B_DIM / M_BLK, NCOL / N_BLK);   // 64 x 256
    const size_t need = WS_TOTAL_HALVES * sizeof(unsigned short); // ~25.2 MB

    if (d_ws != nullptr && ws_size >= need) {
        unsigned short* w = (unsigned short*)d_ws;
        convert_split_kernel<<<(NA_ + NB_) / 1024, 256, 0, stream>>>(x1, x2, w);
        gemm_tdm_kernel<<<grid, 256, 0, stream>>>(w, out);
    } else {
        fused_kernel<<<grid, 256, 0, stream>>>(x1, x2, out);
    }
}
